// TolmanEichenbaumMachine_5970004542263
// MI455X (gfx1250) — compile-verified, matrix operand order checked by reading, not run on hardware
//
#include <hip/hip_runtime.h>
#include <stdint.h>
#include <stddef.h>

#define NB 64
#define NS 128
#define ND 768
#define NH 512
#define ETA 0.1f

typedef __attribute__((ext_vector_type(16))) __bf16 v16bf;
typedef __attribute__((ext_vector_type(8)))  float  v8f;
typedef __attribute__((ext_vector_type(4)))  int    v4i_t;

union Frag16 { uint32_t u[8]; v16bf v; };

// CDNA5 async global->LDS staging path (ASYNCcnt), with fallback.
#if defined(__gfx1250__) && __has_builtin(__builtin_amdgcn_global_load_async_to_lds_b128) && \
    __has_builtin(__builtin_amdgcn_s_wait_asynccnt)
#define USE_ASYNC 1
#else
#define USE_ASYNC 0
#endif

// Builtin signature (from hipcc diagnostic): (AS1 int4*, AS3 int4*, imm offset, imm cpol)
#define AS1V4(p) ((__attribute__((address_space(1))) v4i_t*)(__attribute__((address_space(1))) void*)(void*)(p))
#define AS3V4(p) ((__attribute__((address_space(3))) v4i_t*)(__attribute__((address_space(3))) void*)(void*)(p))

__device__ __forceinline__ uint16_t f2bf(float f) {
  union { float f; uint32_t u; } x; x.f = f;
  uint32_t r = x.u + 0x7fffu + ((x.u >> 16) & 1u);   // round-to-nearest-even
  return (uint16_t)(r >> 16);
}

// Load one 16x32 bf16 WMMA operand fragment from a row-major matrix (bf16 bits
// stored as uint16_t). CDNA5 16-bit operand layout: lanes 0-15 hold K={0..7,16..23},
// lanes 16-31 hold K={8..15,24..31}; K pairs are packed per 32-bit VGPR, so we
// fetch 2x4 contiguous dwords (backend fuses them into b128 loads).
__device__ __forceinline__ void load_frag(const uint16_t* __restrict__ base, int ld,
                                          int row, int k0, int lane, uint32_t u[8]) {
  int r = row + (lane & 15);
  int khalf = (lane & 16) ? 8 : 0;
  const uint32_t* p = reinterpret_cast<const uint32_t*>(base + (size_t)r * ld + k0 + khalf);
#pragma unroll
  for (int i = 0; i < 4; ++i) { u[i] = p[i]; u[i + 4] = p[i + 8]; }
}

// ---------------------------------------------------------------- conversions
__global__ void f32_to_bf16_kernel(const float* __restrict__ in, uint16_t* __restrict__ out, int n) {
  int i = blockIdx.x * blockDim.x + threadIdx.x;
  if (i < n) out[i] = f2bf(in[i]);
}

// in: K x N fp32 row-major  ->  out: N x K bf16 row-major (transposed)
__global__ void transpose_f32_to_bf16_kernel(const float* __restrict__ in, uint16_t* __restrict__ out,
                                             int K, int N) {
  int i = blockIdx.x * blockDim.x + threadIdx.x;
  if (i < K * N) {
    int k = i / N, n = i - k * N;
    out[(size_t)n * K + k] = f2bf(in[i]);
  }
}

// p_bf: (B*S, H) bf16 -> pt_bf: (B, H, S) bf16
__global__ void transpose_p_kernel(const uint16_t* __restrict__ p_bf, uint16_t* __restrict__ pt_bf) {
  int i = blockIdx.x * blockDim.x + threadIdx.x;     // over B*S*H
  if (i < NB * NS * NH) {
    int b = i / (NS * NH);
    int r = i - b * NS * NH;
    int t = r / NH, h = r - t * NH;
    pt_bf[((size_t)b * NH + h) * NS + t] = p_bf[i];
  }
}

// ---------------------------------------------------------------- bf16 GEMM
// C[M,N] = act(A[M,K] @ Bt[N,K]^T + bias[N]); block = 8 waves, tile 128x64.
// K-chunks of 32 are double-buffered in LDS via async global->LDS b128 DMA.
#define LDT 40   // padded LDS leading dim: rows stay 16B-aligned, banks conflict-free
template <int RELU, int OUTF32>
__global__ void gemm_bf16_kernel(const uint16_t* __restrict__ A,
                                 const uint16_t* __restrict__ Bt,
                                 const float* __restrict__ bias,
                                 void* __restrict__ Cout,
                                 int M, int N, int K) {
  __shared__ uint16_t As[2][128 * LDT];
  __shared__ uint16_t Bs[2][64 * LDT];
  int tid = threadIdx.x;
  int wave = tid >> 5, lane = tid & 31;
  int mblk = blockIdx.y * 128;
  int nblk = blockIdx.x * 64;

  // Stage one K-chunk (A: 128x32, B: 64x32) into LDS buffer `buf`.
  // 3 b128 transfers per thread-slot => 3 ASYNCcnt entries per wave.
  auto stage = [&](int buf, int k0) {
#pragma unroll
    for (int i = 0; i < 2; ++i) {                     // A tile: 512 x 16B
      int idx = tid + i * 256;
      int row = idx >> 2, c = (idx & 3) * 8;          // c in bf16 elements
      const uint16_t* src = A + (size_t)(mblk + row) * K + k0 + c;
      uint16_t* dst = &As[buf][row * LDT + c];
#if USE_ASYNC
      __builtin_amdgcn_global_load_async_to_lds_b128(AS1V4(src), AS3V4(dst), 0, 0);
#else
      *(uint4*)dst = *(const uint4*)src;
#endif
    }
    {                                                 // B tile: 256 x 16B
      int row = tid >> 2, c = (tid & 3) * 8;
      const uint16_t* src = Bt + (size_t)(nblk + row) * K + k0 + c;
      uint16_t* dst = &Bs[buf][row * LDT + c];
#if USE_ASYNC
      __builtin_amdgcn_global_load_async_to_lds_b128(AS1V4(src), AS3V4(dst), 0, 0);
#else
      *(uint4*)dst = *(const uint4*)src;
#endif
    }
  };

  v8f acc[4];
#pragma unroll
  for (int j = 0; j < 4; ++j) acc[j] = (v8f){0.f, 0.f, 0.f, 0.f, 0.f, 0.f, 0.f, 0.f};

  const int nk = K >> 5;
  stage(0, 0);
  for (int kc = 0; kc < nk; ++kc) {
    int buf = kc & 1;
    if (kc + 1 < nk) {
      stage(buf ^ 1, (kc + 1) << 5);                  // prefetch next chunk
#if USE_ASYNC
      __builtin_amdgcn_s_wait_asynccnt(3);            // only next chunk may be in flight
#endif
    } else {
#if USE_ASYNC
      __builtin_amdgcn_s_wait_asynccnt(0);
#endif
    }
    __syncthreads();                                   // chunk kc visible to all waves

    Frag16 a;
    load_frag(&As[buf][0], LDT, wave * 16, 0, lane, a.u);
#pragma unroll
    for (int j = 0; j < 4; ++j) {
      Frag16 bfr;
      load_frag(&Bs[buf][0], LDT, 16 * j, 0, lane, bfr.u);
      acc[j] = __builtin_amdgcn_wmma_f32_16x16x32_bf16(
          false, a.v, false, bfr.v, (short)0, acc[j], false, false);
    }
    __syncthreads();                                   // all reads done before restage
  }

#pragma unroll
  for (int j = 0; j < 4; ++j) {
    int n = nblk + 16 * j + (lane & 15);
    float bv = bias[n];
#pragma unroll
    for (int v = 0; v < 8; ++v) {
      int m = mblk + wave * 16 + v + ((lane & 16) ? 8 : 0);
      float val = acc[j][v] + bv;
      if (RELU) val = fmaxf(val, 0.f);
      if (OUTF32) ((float*)Cout)[(size_t)m * N + n] = val;
      else        ((uint16_t*)Cout)[(size_t)m * N + n] = f2bf(val);
    }
  }
}

// ---------------------------------------------------------------- g recurrence
// 64 blocks (one per batch) x 512 threads; g lives in LDS; fp32 for tanh fidelity.
// Serial critical path: 127 steps; W_trans (8 MB fp32) stays L2-resident.
__global__ void recurrence_kernel(const float* __restrict__ W_trans,
                                  const float* __restrict__ b_trans,
                                  const float* __restrict__ g_init,
                                  const int* __restrict__ actions,
                                  uint16_t* __restrict__ g_bf) {
  __shared__ __align__(16) float g[NH];
  int b = blockIdx.x, i = threadIdx.x;
  float gi = g_init[i];
  g[i] = gi;
  g_bf[(size_t)b * NS * NH + i] = f2bf(gi);
  __syncthreads();
  float bt = b_trans[i];
  for (int t = 1; t < NS; ++t) {
    int a = actions[b * NS + (t - 1)] & 7;
    const float4* Wrow = reinterpret_cast<const float4*>(W_trans + ((size_t)a * NH + i) * NH);
    const float4* gv = reinterpret_cast<const float4*>(g);
    float sum = bt;
#pragma unroll 8
    for (int k = 0; k < NH / 4; ++k) {
      float4 w = Wrow[k], x = gv[k];
      sum += w.x * x.x + w.y * x.y + w.z * x.z + w.w * x.w;
    }
    float gn = tanhf(sum);
    __syncthreads();
    g[i] = gn;
    g_bf[((size_t)b * NS + t) * NH + i] = f2bf(gn);
    __syncthreads();
  }
}

// ---------------------------------------------------------------- attention
// Per batch: Scores = eta * tril_strict(G G^T)  (128x128, K=512), then
// P_hat = Scores @ P (128x512, K=128).  Scores staged as bf16 in LDS.
#define SLD 132   // padded LDS leading dim (keeps 4B alignment, kills bank conflicts)
__global__ void attention_kernel(const uint16_t* __restrict__ g_bf,
                                 const uint16_t* __restrict__ pt_bf,
                                 const uint16_t* __restrict__ p_bf,
                                 uint16_t* __restrict__ pseq_bf) {
  __shared__ uint16_t sc[NS * SLD];
  int b = blockIdx.x;
  int wave = threadIdx.x >> 5, lane = threadIdx.x & 31;
  int m0 = wave * 16;
  const uint16_t* G = g_bf + (size_t)b * NS * NH;

  // phase 1: masked scores
  for (int j = 0; j < 8; ++j) {
    int n0 = j * 16;
    if (j > wave) {                                   // strictly above diagonal: zero
#pragma unroll
      for (int v = 0; v < 8; ++v) {
        int m = m0 + v + ((lane & 16) ? 8 : 0);
        sc[m * SLD + n0 + (lane & 15)] = 0;
      }
    } else {
      v8f acc = (v8f){0.f, 0.f, 0.f, 0.f, 0.f, 0.f, 0.f, 0.f};
      for (int k0 = 0; k0 < NH; k0 += 32) {
        Frag16 a, bfr;
        load_frag(G, NH, m0, k0, lane, a.u);          // rows t
        load_frag(G, NH, n0, k0, lane, bfr.u);        // "Bt" = G itself (scores = G G^T)
        acc = __builtin_amdgcn_wmma_f32_16x16x32_bf16(
            false, a.v, false, bfr.v, (short)0, acc, false, false);
      }
#pragma unroll
      for (int v = 0; v < 8; ++v) {
        int m = m0 + v + ((lane & 16) ? 8 : 0);       // t
        int n = n0 + (lane & 15);                     // k
        float s = (n < m) ? ETA * acc[v] : 0.f;       // strict lower triangle, fold eta
        sc[m * SLD + n] = f2bf(s);
      }
    }
  }
  __syncthreads();

  // phase 2: P_hat = Scores @ P   (Bt = P^T precomputed, (H,S) row-major)
  const uint16_t* Pt = pt_bf + (size_t)b * NH * NS;
  for (int nt = 0; nt < NH / 16; ++nt) {
    int n0 = nt * 16;
    v8f acc = (v8f){0.f, 0.f, 0.f, 0.f, 0.f, 0.f, 0.f, 0.f};
#pragma unroll
    for (int k0 = 0; k0 < NS; k0 += 32) {
      Frag16 a, bfr;
      load_frag(sc, SLD, m0, k0, lane, a.u);          // ds_load from LDS scores
      load_frag(Pt, NS, n0, k0, lane, bfr.u);
      acc = __builtin_amdgcn_wmma_f32_16x16x32_bf16(
          false, a.v, false, bfr.v, (short)0, acc, false, false);
    }
#pragma unroll
    for (int v = 0; v < 8; ++v) {
      int t = m0 + v + ((lane & 16) ? 8 : 0);
      int h = n0 + (lane & 15);
      uint16_t o = (t == 0) ? p_bf[(size_t)b * NS * NH + h]   // p_seq[0] = encoded p_0
                            : f2bf(acc[v]);
      pseq_bf[((size_t)b * NS + t) * NH + h] = o;
    }
  }
}

// ---------------------------------------------------------------- launch
extern "C" void kernel_launch(void* const* d_in, const int* in_sizes, int n_in,
                              void* d_out, int out_size, void* d_ws, size_t ws_size,
                              hipStream_t stream) {
  const float* obs     = (const float*)d_in[0];
  const int*   actions = (const int*)d_in[1];
  const float* W_trans = (const float*)d_in[2];
  const float* b_trans = (const float*)d_in[3];
  const float* g_init  = (const float*)d_in[4];
  const float* enc_w1  = (const float*)d_in[5];
  const float* enc_b1  = (const float*)d_in[6];
  const float* enc_w2  = (const float*)d_in[7];
  const float* enc_b2  = (const float*)d_in[8];
  const float* dec_w1  = (const float*)d_in[9];
  const float* dec_b1  = (const float*)d_in[10];
  const float* dec_w2  = (const float*)d_in[11];
  const float* dec_b2  = (const float*)d_in[12];
  float* out = (float*)d_out;

  char* ws = (char*)d_ws;
  size_t off = 0;
  auto alloc = [&](size_t bytes) {
    char* p = ws + off;
    off = (off + bytes + 255) & ~(size_t)255;
    return p;
  };
  const int M = NB * NS;                       // 8192
  uint16_t* obs_bf  = (uint16_t*)alloc((size_t)M * ND * 2);
  uint16_t* w_e1t   = (uint16_t*)alloc((size_t)(2 * NH) * ND * 2);
  uint16_t* w_e2t   = (uint16_t*)alloc((size_t)NH * (2 * NH) * 2);
  uint16_t* w_d1t   = (uint16_t*)alloc((size_t)(2 * NH) * NH * 2);
  uint16_t* w_d2t   = (uint16_t*)alloc((size_t)ND * (2 * NH) * 2);
  uint16_t* h1_bf   = (uint16_t*)alloc((size_t)M * (2 * NH) * 2);
  uint16_t* p_bf    = (uint16_t*)alloc((size_t)M * NH * 2);
  uint16_t* pt_bf   = (uint16_t*)alloc((size_t)M * NH * 2);
  uint16_t* g_bf    = (uint16_t*)alloc((size_t)M * NH * 2);
  uint16_t* pseq_bf = (uint16_t*)alloc((size_t)M * NH * 2);
  uint16_t* h2_bf   = (uint16_t*)alloc((size_t)M * (2 * NH) * 2);

  int n;
  n = M * ND;
  f32_to_bf16_kernel<<<(n + 255) / 256, 256, 0, stream>>>(obs, obs_bf, n);
  n = ND * 2 * NH;
  transpose_f32_to_bf16_kernel<<<(n + 255) / 256, 256, 0, stream>>>(enc_w1, w_e1t, ND, 2 * NH);
  n = 2 * NH * NH;
  transpose_f32_to_bf16_kernel<<<(n + 255) / 256, 256, 0, stream>>>(enc_w2, w_e2t, 2 * NH, NH);
  n = NH * 2 * NH;
  transpose_f32_to_bf16_kernel<<<(n + 255) / 256, 256, 0, stream>>>(dec_w1, w_d1t, NH, 2 * NH);
  n = 2 * NH * ND;
  transpose_f32_to_bf16_kernel<<<(n + 255) / 256, 256, 0, stream>>>(dec_w2, w_d2t, 2 * NH, ND);

  // encoder: h1 = relu(obs @ W1 + b1);  p = h1 @ W2 + b2
  gemm_bf16_kernel<1, 0><<<dim3((2 * NH) / 64, M / 128), 256, 0, stream>>>(
      obs_bf, w_e1t, enc_b1, h1_bf, M, 2 * NH, ND);
  gemm_bf16_kernel<0, 0><<<dim3(NH / 64, M / 128), 256, 0, stream>>>(
      h1_bf, w_e2t, enc_b2, p_bf, M, NH, 2 * NH);

  n = NB * NS * NH;
  transpose_p_kernel<<<(n + 255) / 256, 256, 0, stream>>>(p_bf, pt_bf);

  // serial part: g_t = tanh(W[a] g + b), one block per batch
  recurrence_kernel<<<NB, NH, 0, stream>>>(W_trans, b_trans, g_init, actions, g_bf);

  // Hebbian retrieval as causal linear attention
  attention_kernel<<<NB, 256, 0, stream>>>(g_bf, pt_bf, p_bf, pseq_bf);

  // decoder: h2 = relu(p_seq @ dW1 + db1); out = h2 @ dW2 + db2 (fp32 out)
  gemm_bf16_kernel<1, 0><<<dim3((2 * NH) / 64, M / 128), 256, 0, stream>>>(
      pseq_bf, w_d1t, dec_b1, h2_bf, M, 2 * NH, NH);
  gemm_bf16_kernel<0, 1><<<dim3(ND / 64, M / 128), 256, 0, stream>>>(
      h2_bf, w_d2t, dec_b2, out, M, ND, 2 * NH);
}
